// BiMambaBlock_17214228922851
// MI455X (gfx1250) — compile-verified, matrix operand order checked by reading, not run on hardware
//
#include <hip/hip_runtime.h>
#include <math.h>

#define BATCH     4
#define SEQ       2048
#define DMODEL    768
#define DSTATE    16
#define DCONV     4
#define HEADDIM   64
#define DINNER    1536
#define NHEADS    24
#define CONVDIM   1568   // DINNER + 2*DSTATE
#define DIP       3128   // 2*DINNER + 2*DSTATE + NHEADS
#define DIPPAD    3136   // DIP padded to multiple of 64
#define NROWS     (BATCH*SEQ)   // 8192

typedef __attribute__((ext_vector_type(16))) __bf16      v16bf;
typedef __attribute__((ext_vector_type(8)))  float       v8f;
typedef __attribute__((ext_vector_type(4)))  unsigned int u32x4;
typedef __attribute__((ext_vector_type(4)))  int          i32x4;

union Frag16 { v16bf v; u32x4 q[2]; };

#if __has_builtin(__builtin_amdgcn_global_load_async_to_lds_b128)
#define HAVE_ASYNC_LDS 1
typedef __attribute__((address_space(1))) i32x4 as1_i32x4;   // global int4
typedef __attribute__((address_space(3))) i32x4 as3_i32x4;   // LDS int4
#endif

__device__ __forceinline__ void wait_asynccnt0() {
#if __has_builtin(__builtin_amdgcn_s_wait_asynccnt)
    __builtin_amdgcn_s_wait_asynccnt(0);
#else
    asm volatile("s_wait_asynccnt 0x0" ::: "memory");
#endif
}

__device__ __forceinline__ unsigned short f2bf(float f) {
    union { float f; unsigned u; } x; x.f = f;
    unsigned r = x.u + 0x7FFFu + ((x.u >> 16) & 1u);
    return (unsigned short)(r >> 16);
}
__device__ __forceinline__ float geluf(float v) {
    return 0.5f * v * (1.0f + erff(v * 0.70710678118654752f));
}
__device__ __forceinline__ float siluf(float v) {
    return v / (1.0f + __expf(-v));
}

// ---------------------------------------------------------------------------
// weight prep: f32 B[K,N] -> bf16 B^T[Npad,K] (pad rows zero-filled)
// ---------------------------------------------------------------------------
__global__ void cvt_transpose_kernel(const float* __restrict__ in,
                                     unsigned short* __restrict__ out,
                                     int K, int N, int Npad) {
    long long idx = (long long)blockIdx.x * blockDim.x + threadIdx.x;
    long long total = (long long)Npad * K;
    if (idx >= total) return;
    int n = (int)(idx / K);
    int k = (int)(idx % K);
    out[idx] = (n < N) ? f2bf(in[(size_t)k * N + n]) : (unsigned short)0;
}

// ---------------------------------------------------------------------------
// LayerNorm over DMODEL, bf16 output.  1 block per row, 256 threads.
// ---------------------------------------------------------------------------
__global__ __launch_bounds__(256) void ln1_kernel(const float* __restrict__ x,
                                                  const float* __restrict__ w,
                                                  const float* __restrict__ b,
                                                  unsigned short* __restrict__ out) {
    int row = blockIdx.x, tid = threadIdx.x;
    const float* xr = x + (size_t)row * DMODEL;
    float v[3], s = 0.f, ss = 0.f;
#pragma unroll
    for (int i = 0; i < 3; i++) { v[i] = xr[tid + 256 * i]; s += v[i]; ss += v[i] * v[i]; }
    __shared__ float rs[256], rq[256];
    rs[tid] = s; rq[tid] = ss; __syncthreads();
    for (int off = 128; off > 0; off >>= 1) {
        if (tid < off) { rs[tid] += rs[tid + off]; rq[tid] += rq[tid + off]; }
        __syncthreads();
    }
    float mean = rs[0] * (1.f / DMODEL);
    float var  = rq[0] * (1.f / DMODEL) - mean * mean;
    float inv  = rsqrtf(var + 1e-5f);
#pragma unroll
    for (int i = 0; i < 3; i++) {
        int c = tid + 256 * i;
        out[(size_t)row * DMODEL + c] = f2bf((v[i] - mean) * inv * w[c] + b[c]);
    }
}

// ---------------------------------------------------------------------------
// bf16 WMMA GEMM:  C[M,N] = A[M,K] * B[K,N], B supplied pre-transposed/padded
// as BT[Npad,K] (row-major).  Block tile 128x64, BK=32, 256 threads = 8 waves,
// double-buffered LDS; async memory->LDS copies when the toolchain has them.
// mode: bit0 = bf16 output, bit1 = bias+GELU epilogue.
// store at Cout[row*ldc + coloff + col], cols >= N masked.
// ---------------------------------------------------------------------------
#define GLDA 40
__global__ __launch_bounds__(256) void gemm_bf16_kernel(
        const unsigned short* __restrict__ A,
        const unsigned short* __restrict__ BT,
        void* __restrict__ Cout,
        const float* __restrict__ bias,
        int M, int N, int K, int ldc, int coloff, int mode) {
    __shared__ unsigned short As[2][128 * GLDA];
    __shared__ unsigned short Bs[2][64 * GLDA];

    int tid  = threadIdx.x;
    int n0   = blockIdx.x * 64;
    int m0   = blockIdx.y * 128;
    int wave = tid >> 5, lane = tid & 31;
    int wm = wave & 3, wn = wave >> 2;
    int mloc = lane & 15, khalf = (lane >> 4) & 1;

    // staging geometry (all 16B aligned, contiguous copies)
    int arow0 = (tid + 0)   >> 2, ako0 = ((tid + 0)   & 3) * 8;
    int arow1 = (tid + 256) >> 2, ako1 = ((tid + 256) & 3) * 8;
    int brow  = tid >> 2,         bko  = (tid & 3) * 8;
    const unsigned short* gA0 = A  + (size_t)(m0 + arow0) * K + ako0;
    const unsigned short* gA1 = A  + (size_t)(m0 + arow1) * K + ako1;
    const unsigned short* gB  = BT + (size_t)(n0 + brow)  * K + bko;

    v8f zero = {0.f, 0.f, 0.f, 0.f, 0.f, 0.f, 0.f, 0.f};
    v8f acc[2][2];
    acc[0][0] = zero; acc[0][1] = zero; acc[1][0] = zero; acc[1][1] = zero;

    int ntiles = K >> 5;

#ifdef HAVE_ASYNC_LDS
    auto stage = [&](int k0, int buf) {
        __builtin_amdgcn_global_load_async_to_lds_b128(
            (as1_i32x4*)(gA0 + k0), (as3_i32x4*)&As[buf][arow0 * GLDA + ako0], 0, 0);
        __builtin_amdgcn_global_load_async_to_lds_b128(
            (as1_i32x4*)(gA1 + k0), (as3_i32x4*)&As[buf][arow1 * GLDA + ako1], 0, 0);
        __builtin_amdgcn_global_load_async_to_lds_b128(
            (as1_i32x4*)(gB + k0),  (as3_i32x4*)&Bs[buf][brow * GLDA + bko],   0, 0);
    };
    stage(0, 0);
    wait_asynccnt0();
#else
    u32x4 ra0, ra1, rb;
    auto load_regs = [&](int k0) {
        ra0 = *(const u32x4*)(gA0 + k0);
        ra1 = *(const u32x4*)(gA1 + k0);
        rb  = *(const u32x4*)(gB + k0);
    };
    auto store_regs = [&](int buf) {
        *(u32x4*)(&As[buf][arow0 * GLDA + ako0]) = ra0;
        *(u32x4*)(&As[buf][arow1 * GLDA + ako1]) = ra1;
        *(u32x4*)(&Bs[buf][brow * GLDA + bko])   = rb;
    };
    load_regs(0);
    store_regs(0);
#endif
    __syncthreads();

    for (int it = 0; it < ntiles; ++it) {
        int cur = it & 1, nxt = cur ^ 1;
#ifdef HAVE_ASYNC_LDS
        if (it + 1 < ntiles) stage((it + 1) << 5, nxt);
#else
        if (it + 1 < ntiles) load_regs((it + 1) << 5);
#endif
        // --- fragments: e0..7 <-> k = 8*khalf+0..7, e8..15 <-> k = 16+8*khalf+0..7
        Frag16 af[2], bf[2];
#pragma unroll
        for (int mt = 0; mt < 2; mt++) {
            int ar = wm * 32 + mt * 16 + mloc;
            af[mt].q[0] = *(const u32x4*)(&As[cur][ar * GLDA + 8 * khalf]);
            af[mt].q[1] = *(const u32x4*)(&As[cur][ar * GLDA + 16 + 8 * khalf]);
        }
#pragma unroll
        for (int nt = 0; nt < 2; nt++) {
            int bn = wn * 32 + nt * 16 + mloc;
            bf[nt].q[0] = *(const u32x4*)(&Bs[cur][bn * GLDA + 8 * khalf]);
            bf[nt].q[1] = *(const u32x4*)(&Bs[cur][bn * GLDA + 16 + 8 * khalf]);
        }
#pragma unroll
        for (int mt = 0; mt < 2; mt++)
#pragma unroll
            for (int nt = 0; nt < 2; nt++)
                acc[mt][nt] = __builtin_amdgcn_wmma_f32_16x16x32_bf16(
                    false, af[mt].v, false, bf[nt].v,
                    (short)0, acc[mt][nt], false, false);

        if (it + 1 < ntiles) {
#ifdef HAVE_ASYNC_LDS
            wait_asynccnt0();
#else
            store_regs(nxt);
#endif
            __syncthreads();
        }
    }

    // --- epilogue.  C/D layout: VGPR r -> M = r + 8*(lane>>4), N = lane&15
    int rbase = (lane >> 4) * 8;
#pragma unroll
    for (int mt = 0; mt < 2; mt++)
#pragma unroll
        for (int nt = 0; nt < 2; nt++) {
            int col = n0 + wn * 32 + nt * 16 + mloc;
            if (col < N) {
#pragma unroll
                for (int r = 0; r < 8; r++) {
                    int row = m0 + wm * 32 + mt * 16 + rbase + r;
                    float v = acc[mt][nt][r];
                    if (mode & 2) { v = geluf(v + bias[col]); }
                    size_t off = (size_t)row * ldc + coloff + col;
                    if (mode & 1) ((unsigned short*)Cout)[off] = f2bf(v);
                    else          ((float*)Cout)[off]          = v;
                }
            }
        }
}

// ---------------------------------------------------------------------------
// depthwise causal conv (width 4) + SiLU.  xc[b,s,c], dir flips token index.
// ---------------------------------------------------------------------------
__global__ void conv_kernel(const float* __restrict__ zx,
                            const float* __restrict__ cw,
                            const float* __restrict__ cb,
                            float* __restrict__ xc, int dir) {
    long long idx = (long long)blockIdx.x * blockDim.x + threadIdx.x;
    long long total = (long long)BATCH * SEQ * CONVDIM;
    if (idx >= total) return;
    int c = (int)(idx % CONVDIM);
    int s = (int)((idx / CONVDIM) % SEQ);
    int b = (int)(idx / ((long long)CONVDIM * SEQ));
    float acc = cb[c];
#pragma unroll
    for (int k = 0; k < DCONV; k++) {
        int sp = s + k - (DCONV - 1);
        if (sp >= 0) {
            int tok = dir ? (SEQ - 1 - sp) : sp;
            acc += cw[c * DCONV + k] * zx[((size_t)b * SEQ + tok) * DIP + DINNER + c];
        }
    }
    xc[((size_t)b * SEQ + s) * CONVDIM + c] = siluf(acc);
}

// ---------------------------------------------------------------------------
// dt = softplus(raw + dt_bias),  dA = exp(dt * (-exp(A_log)))   (s-indexed)
// ---------------------------------------------------------------------------
__global__ void dtda_kernel(const float* __restrict__ zx,
                            const float* __restrict__ dt_bias,
                            const float* __restrict__ A_log,
                            float* __restrict__ dt, float* __restrict__ dA,
                            int dir) {
    int idx = blockIdx.x * blockDim.x + threadIdx.x;
    int total = BATCH * SEQ * NHEADS;
    if (idx >= total) return;
    int h = idx % NHEADS;
    int s = (idx / NHEADS) % SEQ;
    int b = idx / (NHEADS * SEQ);
    int tok = dir ? (SEQ - 1 - s) : s;
    float raw = zx[((size_t)b * SEQ + tok) * DIP + (DIP - NHEADS) + h] + dt_bias[h];
    float d   = (raw > 20.f) ? raw : log1pf(__expf(raw));
    float a   = -__expf(A_log[h]);
    size_t o  = ((size_t)b * SEQ + s) * NHEADS + h;
    dt[o] = d;
    dA[o] = __expf(d * a);
}

// ---------------------------------------------------------------------------
// selective scan: 1 block per (batch,head), 128 threads.
// thread owns p = tid>>1, states n in [(tid&1)*8, +8).  Chunked LDS staging.
// y written in ORIGINAL token order (includes D*x term).
// ---------------------------------------------------------------------------
#define CHUNK 32
__global__ __launch_bounds__(128) void scan_kernel(const float* __restrict__ xc,
                                                   const float* __restrict__ dt,
                                                   const float* __restrict__ dA,
                                                   const float* __restrict__ Dp,
                                                   float* __restrict__ y, int dir) {
    int bh = blockIdx.x;
    int b = bh / NHEADS, h = bh % NHEADS;
    int tid = threadIdx.x;
    int p  = tid >> 1;
    int nb = (tid & 1) * 8;

    __shared__ float xs[CHUNK][HEADDIM];
    __shared__ float Bsh[CHUNK][DSTATE];
    __shared__ float Csh[CHUNK][DSTATE];
    __shared__ float dAs[CHUNK], dts[CHUNK];

    float hreg[8];
#pragma unroll
    for (int j = 0; j < 8; j++) hreg[j] = 0.f;
    float Dh = Dp[h];
    const float* xcb = xc + (size_t)b * SEQ * CONVDIM;

    for (int c0 = 0; c0 < SEQ; c0 += CHUNK) {
        __syncthreads();
        for (int i = tid; i < CHUNK * HEADDIM; i += 128) {
            int t = i >> 6, pp = i & 63;
            xs[t][pp] = xcb[(size_t)(c0 + t) * CONVDIM + h * HEADDIM + pp];
        }
        for (int i = tid; i < CHUNK * DSTATE; i += 128) {
            int t = i >> 4, n = i & 15;
            Bsh[t][n] = xcb[(size_t)(c0 + t) * CONVDIM + DINNER + n];
            Csh[t][n] = xcb[(size_t)(c0 + t) * CONVDIM + DINNER + DSTATE + n];
        }
        if (tid < CHUNK) {
            size_t o = ((size_t)b * SEQ + c0 + tid) * NHEADS + h;
            dAs[tid] = dA[o];
            dts[tid] = dt[o];
        }
        __syncthreads();
#pragma unroll 4
        for (int t = 0; t < CHUNK; t++) {
            float a  = dAs[t];
            float xv = xs[t][p];
            float dx = dts[t] * xv;
            float part = 0.f;
#pragma unroll
            for (int j = 0; j < 8; j++) {
                hreg[j] = hreg[j] * a + dx * Bsh[t][nb + j];
                part   += hreg[j] * Csh[t][nb + j];
            }
            part += __shfl_xor(part, 1, 32);
            if ((tid & 1) == 0) {
                int s   = c0 + t;
                int tok = dir ? (SEQ - 1 - s) : s;
                y[((size_t)b * SEQ + tok) * DINNER + h * HEADDIM + p] = part + Dh * xv;
            }
        }
    }
}

// ---------------------------------------------------------------------------
// gated RMSNorm: yz = y*silu(z); yz *= rsqrt(mean(yz^2)+eps)*norm_w -> bf16
// z comes from zx columns [0,DINNER), same token order as y.
// ---------------------------------------------------------------------------
__global__ __launch_bounds__(256) void gated_rms_kernel(const float* __restrict__ y,
                                                        const float* __restrict__ zx,
                                                        const float* __restrict__ nw,
                                                        unsigned short* __restrict__ out) {
    int row = blockIdx.x, tid = threadIdx.x;
    const float* yr = y  + (size_t)row * DINNER;
    const float* zr = zx + (size_t)row * DIP;
    float v[6], ss = 0.f;
#pragma unroll
    for (int i = 0; i < 6; i++) {
        int c = tid + 256 * i;
        v[i] = yr[c] * siluf(zr[c]);
        ss  += v[i] * v[i];
    }
    __shared__ float rq[256];
    rq[tid] = ss; __syncthreads();
    for (int off = 128; off > 0; off >>= 1) {
        if (tid < off) rq[tid] += rq[tid + off];
        __syncthreads();
    }
    float inv = rsqrtf(rq[0] * (1.f / DINNER) + 1e-5f);
#pragma unroll
    for (int i = 0; i < 6; i++) {
        int c = tid + 256 * i;
        out[(size_t)row * DINNER + c] = f2bf(v[i] * inv * nw[c]);
    }
}

// ---------------------------------------------------------------------------
// final: LayerNorm(x + h2) -> d_out (f32)
// ---------------------------------------------------------------------------
__global__ __launch_bounds__(256) void final_ln_kernel(const float* __restrict__ x,
                                                       const float* __restrict__ h2,
                                                       const float* __restrict__ w,
                                                       const float* __restrict__ b,
                                                       float* __restrict__ out) {
    int row = blockIdx.x, tid = threadIdx.x;
    const float* xr = x  + (size_t)row * DMODEL;
    const float* hr = h2 + (size_t)row * DMODEL;
    float v[3], s = 0.f, ss = 0.f;
#pragma unroll
    for (int i = 0; i < 3; i++) {
        int c = tid + 256 * i;
        v[i] = xr[c] + hr[c];
        s += v[i]; ss += v[i] * v[i];
    }
    __shared__ float rs[256], rq[256];
    rs[tid] = s; rq[tid] = ss; __syncthreads();
    for (int off = 128; off > 0; off >>= 1) {
        if (tid < off) { rs[tid] += rs[tid + off]; rq[tid] += rq[tid + off]; }
        __syncthreads();
    }
    float mean = rs[0] * (1.f / DMODEL);
    float var  = rq[0] * (1.f / DMODEL) - mean * mean;
    float inv  = rsqrtf(var + 1e-5f);
#pragma unroll
    for (int i = 0; i < 3; i++) {
        int c = tid + 256 * i;
        out[(size_t)row * DMODEL + c] = (v[i] - mean) * inv * w[c] + b[c];
    }
}

// ---------------------------------------------------------------------------
extern "C" void kernel_launch(void* const* d_in, const int* in_sizes, int n_in,
                              void* d_out, int out_size, void* d_ws, size_t ws_size,
                              hipStream_t stream) {
    // ---- inputs (setup_inputs dict order) ----
    const float* x       = (const float*)d_in[0];
    const float* ln1_w   = (const float*)d_in[1];
    const float* ln1_b   = (const float*)d_in[2];
    const float* ln2_w   = (const float*)d_in[3];
    const float* ln2_b   = (const float*)d_in[4];
    const float* ffn_w1  = (const float*)d_in[5];
    const float* ffn_b1  = (const float*)d_in[6];
    const float* ffn_w2  = (const float*)d_in[7];
    const float* ffn_b2  = (const float*)d_in[8];
    const float* in_w[2]    = { (const float*)d_in[9],  (const float*)d_in[17] };
    const float* conv_w[2]  = { (const float*)d_in[10], (const float*)d_in[18] };
    const float* conv_b[2]  = { (const float*)d_in[11], (const float*)d_in[19] };
    const float* dt_bias[2] = { (const float*)d_in[12], (const float*)d_in[20] };
    const float* A_log[2]   = { (const float*)d_in[13], (const float*)d_in[21] };
    const float* Dp[2]      = { (const float*)d_in[14], (const float*)d_in[22] };
    const float* norm_w[2]  = { (const float*)d_in[15], (const float*)d_in[23] };
    const float* out_w[2]   = { (const float*)d_in[16], (const float*)d_in[24] };
    float* out = (float*)d_out;

    // ---- workspace carve-up (256B aligned) ----
    size_t off = 0;
    auto alloc = [&](size_t bytes) -> char* {
        char* p = (char*)d_ws + off;
        off += (bytes + 255) & ~(size_t)255;
        return p;
    };
    unsigned short* xn16   = (unsigned short*)alloc((size_t)NROWS * DMODEL * 2);
    // all weights stored transposed: BT[Npad, K]
    unsigned short* inw16t[2]  = { (unsigned short*)alloc((size_t)DIPPAD * DMODEL * 2),
                                   (unsigned short*)alloc((size_t)DIPPAD * DMODEL * 2) };
    unsigned short* outw16t[2] = { (unsigned short*)alloc((size_t)DMODEL * DINNER * 2),
                                   (unsigned short*)alloc((size_t)DMODEL * DINNER * 2) };
    unsigned short* w1_16t = (unsigned short*)alloc((size_t)3072 * DINNER * 2);
    unsigned short* w2_16t = (unsigned short*)alloc((size_t)DMODEL * 3072 * 2);
    float* zx  = (float*)alloc((size_t)NROWS * DIP * 4);       // reused per dir
    float* xc  = (float*)alloc((size_t)NROWS * CONVDIM * 4);   // reused per dir
    float* dt  = (float*)alloc((size_t)NROWS * NHEADS * 4);
    float* dA  = (float*)alloc((size_t)NROWS * NHEADS * 4);
    float* yb  = (float*)alloc((size_t)NROWS * DINNER * 4);    // reused per dir
    unsigned short* yn16[2] = { (unsigned short*)alloc((size_t)NROWS * DINNER * 2),
                                (unsigned short*)alloc((size_t)NROWS * DINNER * 2) };
    unsigned short* bi16  = (unsigned short*)alloc((size_t)NROWS * DINNER * 2);
    unsigned short* h1_16 = (unsigned short*)alloc((size_t)NROWS * 3072 * 2);
    float* h2 = (float*)alloc((size_t)NROWS * DMODEL * 4);
    (void)ws_size; (void)n_in; (void)in_sizes; (void)out_size;

    auto cvt_t = [&](const float* src, unsigned short* dst, int K, int N, int Npad) {
        long long tot = (long long)Npad * K;
        cvt_transpose_kernel<<<(unsigned)((tot + 255) / 256), 256, 0, stream>>>(
            src, dst, K, N, Npad);
    };
    // weights -> bf16 transposed (+pad)
    cvt_t(in_w[0],  inw16t[0],  DMODEL, DIP,    DIPPAD);
    cvt_t(in_w[1],  inw16t[1],  DMODEL, DIP,    DIPPAD);
    cvt_t(out_w[0], outw16t[0], DINNER, DMODEL, DMODEL);
    cvt_t(out_w[1], outw16t[1], DINNER, DMODEL, DMODEL);
    cvt_t(ffn_w1,   w1_16t,     DINNER, 3072,   3072);
    cvt_t(ffn_w2,   w2_16t,     3072,   DMODEL, DMODEL);

    // LayerNorm1 -> bf16
    ln1_kernel<<<NROWS, 256, 0, stream>>>(x, ln1_w, ln1_b, xn16);

    const long long conv_total = (long long)BATCH * SEQ * CONVDIM;
    const int dtda_total = BATCH * SEQ * NHEADS;

    for (int dir = 0; dir < 2; dir++) {
        // in_proj: [8192 x 768] x [768 x 3128] -> zx f32
        gemm_bf16_kernel<<<dim3(DIPPAD / 64, NROWS / 128), 256, 0, stream>>>(
            xn16, inw16t[dir], zx, nullptr, NROWS, DIP, DMODEL, DIP, 0, /*mode=*/0);
        // conv + silu (dir-flipped indexing)
        conv_kernel<<<(unsigned)((conv_total + 255) / 256), 256, 0, stream>>>(
            zx, conv_w[dir], conv_b[dir], xc, dir);
        // dt / dA
        dtda_kernel<<<(dtda_total + 255) / 256, 256, 0, stream>>>(
            zx, dt_bias[dir], A_log[dir], dt, dA, dir);
        // selective scan (writes y in original token order, includes D*x)
        scan_kernel<<<BATCH * NHEADS, 128, 0, stream>>>(xc, dt, dA, Dp[dir], yb, dir);
        // gated RMSNorm -> bf16 (z read from zx, token order)
        gated_rms_kernel<<<NROWS, 256, 0, stream>>>(yb, zx, norm_w[dir], yn16[dir]);
    }

    // out_proj -> concatenated bi (bf16), ldc=1536, column offsets 0 / 768
    gemm_bf16_kernel<<<dim3(DMODEL / 64, NROWS / 128), 256, 0, stream>>>(
        yn16[0], outw16t[0], bi16, nullptr, NROWS, DMODEL, DINNER, DINNER, 0,   /*mode=*/1);
    gemm_bf16_kernel<<<dim3(DMODEL / 64, NROWS / 128), 256, 0, stream>>>(
        yn16[1], outw16t[1], bi16, nullptr, NROWS, DMODEL, DINNER, DINNER, 768, /*mode=*/1);

    // FFN1: [8192 x 1536] x [1536 x 3072], fused bias+GELU -> bf16
    gemm_bf16_kernel<<<dim3(3072 / 64, NROWS / 128), 256, 0, stream>>>(
        bi16, w1_16t, h1_16, ffn_b1, NROWS, 3072, DINNER, 3072, 0, /*mode=*/3);
    // FFN2: [8192 x 3072] x [3072 x 768], fused bias+GELU -> f32
    gemm_bf16_kernel<<<dim3(DMODEL / 64, NROWS / 128), 256, 0, stream>>>(
        h1_16, w2_16t, h2, ffn_b2, NROWS, DMODEL, 3072, DMODEL, 0, /*mode=*/2);

    // residual + LayerNorm2 -> out
    final_ln_kernel<<<NROWS, 256, 0, stream>>>(x, h2, ln2_w, ln2_b, out);
}
